// LongTermGNN_34162169872949
// MI455X (gfx1250) — matrix operand order repaired and checked by reading, not compiled, not run on hardware
//
#include <hip/hip_runtime.h>

#define N_NODES 50000
#define N_EDGES 800000
#define DIM     128
#define NLAYERS 2
#define NREL    8
#define NBASIS  4
#define WCOLS   (NBASIS * DIM + DIM)   // 640
#define NEG_SLOPE 0.2f
#define LN_EPS    1e-5f

typedef __attribute__((ext_vector_type(16))) _Float16 v16h;
typedef __attribute__((ext_vector_type(8)))  _Float16 v8h;
typedef __attribute__((ext_vector_type(8)))  float    v8f;

// order-preserving float<->uint encoding for atomic max on signed floats
__device__ __forceinline__ unsigned enc_f32(float f) {
    unsigned u = __float_as_uint(f);
    return (u & 0x80000000u) ? ~u : (u | 0x80000000u);
}
__device__ __forceinline__ float dec_f32(unsigned u) {
    return (u & 0x80000000u) ? __uint_as_float(u ^ 0x80000000u)
                             : __uint_as_float(~u);
}

__device__ __forceinline__ float warp_sum32(float v) {
    v += __shfl_xor(v, 16, 32);
    v += __shfl_xor(v, 8, 32);
    v += __shfl_xor(v, 4, 32);
    v += __shfl_xor(v, 2, 32);
    v += __shfl_xor(v, 1, 32);
    return v;
}

// ---------------- conversion kernels ----------------

__global__ void k_f32_to_f16(const float* __restrict__ src,
                             _Float16* __restrict__ dst, int n) {
    int i = blockIdx.x * blockDim.x + threadIdx.x;
    if (i < n) dst[i] = (_Float16)src[i];
}

// pack [basis_l (B,D,D) | root_l (D,D)] into column-major f16: wT[c][k] = W[k][c]
__global__ void k_pack_weights(const float* __restrict__ basis_l,
                               const float* __restrict__ root_l,
                               _Float16* __restrict__ wT) {
    int i = blockIdx.x * blockDim.x + threadIdx.x;
    if (i >= WCOLS * DIM) return;
    int c = i / DIM, k = i % DIM;
    float w;
    if (c < NBASIS * DIM) {
        int b = c / DIM, o = c % DIM;
        w = basis_l[(b * DIM + k) * DIM + o];
    } else {
        int o = c - NBASIS * DIM;
        w = root_l[k * DIM + o];
    }
    wT[c * DIM + k] = (_Float16)w;
}

// ---------------- WMMA GEMM: [N,128] x [128,640] -> x_basis [N,512], x_root [N,128] ----------------

__global__ void __launch_bounds__(256)
k_gemm_wmma(const _Float16* __restrict__ xh,    // [N,128] row-major
            const _Float16* __restrict__ wT,    // [640,128]  (col-major weight)
            float* __restrict__ x_basis,        // [N,512]
            float* __restrict__ x_root) {       // [N,128]
    const int COL_TILES = WCOLS / 16;                       // 40
    const int TILES = (N_NODES / 16) * COL_TILES;           // 3125*40 = 125000
    int warp = (int)((blockIdx.x * blockDim.x + threadIdx.x) >> 5);
    if (warp >= TILES) return;
    int lane = threadIdx.x & 31;
    int tm = warp / COL_TILES, tn = warp % COL_TILES;
    int m0 = tm * 16, n0 = tn * 16;
    int half16 = (lane >> 4) & 1;                           // lanes 16..31
    const _Float16* arow = xh + (size_t)(m0 + (lane & 15)) * DIM;
    const _Float16* bcol = wT + (size_t)(n0 + (lane & 15)) * DIM;

    v8f acc = {};
    #pragma unroll
    for (int kb = 0; kb < DIM / 32; ++kb) {
        // A frag (16-bit 16x32 layout): lanes<16 hold k=0..7,16..23; lanes>=16 hold k=8..15,24..31
        int ab = kb * 32 + half16 * 8;
        v8h alo = *(const v8h*)(arow + ab);
        v8h ahi = *(const v8h*)(arow + ab + 16);
        v16h a = __builtin_shufflevector(alo, ahi, 0,1,2,3,4,5,6,7,8,9,10,11,12,13,14,15);
        // B frag (32x16): lanes<16 hold k=0..7,16..23; lanes>=16 hold k=8..15,24..31
        int bb = kb * 32 + half16 * 8;
        v8h blo = *(const v8h*)(bcol + bb);
        v8h bhi = *(const v8h*)(bcol + bb + 16);
        v16h b = __builtin_shufflevector(blo, bhi, 0,1,2,3,4,5,6,7,8,9,10,11,12,13,14,15);
        acc = __builtin_amdgcn_wmma_f32_16x16x32_f16(false, a, false, b, (short)0, acc,
                                                     false, false);
    }

    // C/D layout: VGPR j -> row m0 + j (+8 for lanes>=16), col n0 + lane%16
    int rbase = m0 + half16 * 8;
    int c = n0 + (lane & 15);
    if (n0 < NBASIS * DIM) {
        float* o = x_basis + (size_t)rbase * (NBASIS * DIM) + c;
        #pragma unroll
        for (int j = 0; j < 8; ++j) o[(size_t)j * (NBASIS * DIM)] = acc[j];
    } else {
        float* o = x_root + (size_t)rbase * DIM + (c - NBASIS * DIM);
        #pragma unroll
        for (int j = 0; j < 8; ++j) o[(size_t)j * DIM] = acc[j];
    }
}

// ---------------- per-layer init ----------------

__global__ void k_init(float* __restrict__ aggr, unsigned* __restrict__ nmax,
                       float* __restrict__ nsum) {
    int i = blockIdx.x * blockDim.x + threadIdx.x;
    if (i < N_NODES * DIM) aggr[i] = 0.f;
    if (i < N_NODES) { nmax[i] = 0x007FFFFFu; /* enc(-inf) */ nsum[i] = 0.f; }
}

// ---------------- edge pass A: logits + segment max ----------------

__global__ void __launch_bounds__(256)
k_edge_logits(const int* __restrict__ ei, const int* __restrict__ et,
              const float* __restrict__ att_r_l,   // [R,B]
              const float* __restrict__ att_l,     // [256]
              const float* __restrict__ x_basis, const float* __restrict__ x_root,
              float* __restrict__ alpha_raw, unsigned* __restrict__ nmax) {
    int e = (int)((blockIdx.x * blockDim.x + threadIdx.x) >> 5);
    if (e >= N_EDGES) return;
    int lane = threadIdx.x & 31;
    int s = ei[e], d = ei[N_EDGES + e], t = et[e];
    const float* cf = att_r_l + t * NBASIS;
    float c0 = cf[0], c1 = cf[1], c2 = cf[2], c3 = cf[3];
    const float4* xb = (const float4*)(x_basis + (size_t)s * (NBASIS * DIM));
    float4 b0 = xb[lane], b1 = xb[32 + lane], b2 = xb[64 + lane], b3 = xb[96 + lane];
    float4 xj;
    xj.x = c0*b0.x + c1*b1.x + c2*b2.x + c3*b3.x;
    xj.y = c0*b0.y + c1*b1.y + c2*b2.y + c3*b3.y;
    xj.z = c0*b0.z + c1*b1.z + c2*b2.z + c3*b3.z;
    xj.w = c0*b0.w + c1*b1.w + c2*b2.w + c3*b3.w;
    float4 xr = ((const float4*)(x_root + (size_t)d * DIM))[lane];
    float4 ai = ((const float4*)(att_l))[lane];
    float4 aj = ((const float4*)(att_l + DIM))[lane];
    float p = xr.x*ai.x + xr.y*ai.y + xr.z*ai.z + xr.w*ai.w
            + xj.x*aj.x + xj.y*aj.y + xj.z*aj.z + xj.w*aj.w;
    p = warp_sum32(p);
    if (lane == 0) {
        float a = p > 0.f ? p : NEG_SLOPE * p;
        alpha_raw[e] = a;
        atomicMax(nmax + d, enc_f32(a));
    }
}

// ---------------- edge pass B: exp, segment sum, unnormalized aggregate ----------------

__global__ void __launch_bounds__(256)
k_edge_aggr(const int* __restrict__ ei, const int* __restrict__ et,
            const float* __restrict__ att_r_l,
            const float* __restrict__ x_basis,
            const float* __restrict__ alpha_raw,
            const unsigned* __restrict__ nmax,
            float* __restrict__ nsum, float* __restrict__ aggr) {
    int e = (int)((blockIdx.x * blockDim.x + threadIdx.x) >> 5);
    if (e >= N_EDGES) return;
    int lane = threadIdx.x & 31;
    int s = ei[e], d = ei[N_EDGES + e], t = et[e];
    const float* cf = att_r_l + t * NBASIS;
    float c0 = cf[0], c1 = cf[1], c2 = cf[2], c3 = cf[3];
    const float4* xb = (const float4*)(x_basis + (size_t)s * (NBASIS * DIM));
    float4 b0 = xb[lane], b1 = xb[32 + lane], b2 = xb[64 + lane], b3 = xb[96 + lane];
    float4 xj;
    xj.x = c0*b0.x + c1*b1.x + c2*b2.x + c3*b3.x;
    xj.y = c0*b0.y + c1*b1.y + c2*b2.y + c3*b3.y;
    xj.z = c0*b0.z + c1*b1.z + c2*b2.z + c3*b3.z;
    xj.w = c0*b0.w + c1*b1.w + c2*b2.w + c3*b3.w;
    float m = dec_f32(nmax[d]);
    float ee = __expf(alpha_raw[e] - m);
    if (lane == 0) atomicAdd(nsum + d, ee);
    float* ag = aggr + (size_t)d * DIM + lane * 4;
    atomicAdd(ag + 0, ee * xj.x);
    atomicAdd(ag + 1, ee * xj.y);
    atomicAdd(ag + 2, ee * xj.z);
    atomicAdd(ag + 3, ee * xj.w);
}

// ---------------- node finalize: softmax-normalize + residual + LN + tanh ----------------

__global__ void __launch_bounds__(256)
k_node_final(const float* __restrict__ aggr, const float* __restrict__ nsum,
             const float* __restrict__ x_root,
             const float* __restrict__ bias_l, const float* __restrict__ g_l,
             const float* __restrict__ b_l,
             float* __restrict__ out_state, float* __restrict__ out_final) {
    int n = (int)((blockIdx.x * blockDim.x + threadIdx.x) >> 5);
    if (n >= N_NODES) return;
    int lane = threadIdx.x & 31;
    float inv_s = 1.f / (nsum[n] + 1e-16f);
    float4 ag = ((const float4*)(aggr   + (size_t)n * DIM))[lane];
    float4 xr = ((const float4*)(x_root + (size_t)n * DIM))[lane];
    float4 bs = ((const float4*)bias_l)[lane];
    float4 o;
    o.x = ag.x * inv_s + xr.x + bs.x;
    o.y = ag.y * inv_s + xr.y + bs.y;
    o.z = ag.z * inv_s + xr.z + bs.z;
    o.w = ag.w * inv_s + xr.w + bs.w;
    float ssum = o.x + o.y + o.z + o.w;
    float ssq  = o.x*o.x + o.y*o.y + o.z*o.z + o.w*o.w;
    ssum = warp_sum32(ssum);
    ssq  = warp_sum32(ssq);
    float mu  = ssum * (1.f / DIM);
    float var = ssq * (1.f / DIM) - mu * mu;
    float inv = rsqrtf(var + LN_EPS);
    float4 g  = ((const float4*)g_l)[lane];
    float4 bb = ((const float4*)b_l)[lane];
    float4 y;
    y.x = tanhf(g.x * (o.x - mu) * inv + bb.x);
    y.y = tanhf(g.y * (o.y - mu) * inv + bb.y);
    y.z = tanhf(g.z * (o.z - mu) * inv + bb.z);
    y.w = tanhf(g.w * (o.w - mu) * inv + bb.w);
    ((float4*)(out_state + (size_t)n * DIM))[lane] = y;
    if (out_final) ((float4*)(out_final + (size_t)n * DIM))[lane] = y;
}

// ---------------- driver ----------------

extern "C" void kernel_launch(void* const* d_in, const int* in_sizes, int n_in,
                              void* d_out, int out_size, void* d_ws, size_t ws_size,
                              hipStream_t stream) {
    (void)in_sizes; (void)n_in; (void)out_size; (void)ws_size;
    const float* x_in  = (const float*)d_in[0];
    const int*   ei    = (const int*)  d_in[1];
    const int*   et    = (const int*)  d_in[2];
    const float* basis = (const float*)d_in[3];   // [L,B,D,D]
    const float* att_r = (const float*)d_in[4];   // [L,R,B]
    const float* att   = (const float*)d_in[5];   // [L,2D]
    const float* root  = (const float*)d_in[6];   // [L,D,D]
    const float* bias  = (const float*)d_in[7];   // [L,D]
    const float* ln_g  = (const float*)d_in[8];   // [L,D]
    const float* ln_b  = (const float*)d_in[9];   // [L,D]
    float* out = (float*)d_out;                   // [N*D + L*N*D]

    char* ws = (char*)d_ws;
    auto alloc = [&](size_t bytes) -> char* {
        char* p = ws; ws += (bytes + 255) & ~(size_t)255; return p;
    };
    _Float16* xh     = (_Float16*)alloc(sizeof(_Float16) * (size_t)N_NODES * DIM);
    _Float16* wT     = (_Float16*)alloc(sizeof(_Float16) * (size_t)WCOLS * DIM);
    float* x_basis   = (float*)   alloc(sizeof(float) * (size_t)N_NODES * NBASIS * DIM);
    float* x_root    = (float*)   alloc(sizeof(float) * (size_t)N_NODES * DIM);
    float* alpha     = (float*)   alloc(sizeof(float) * (size_t)N_EDGES);
    unsigned* nmax   = (unsigned*)alloc(sizeof(unsigned) * (size_t)N_NODES);
    float* nsum      = (float*)   alloc(sizeof(float) * (size_t)N_NODES);
    float* aggr      = (float*)   alloc(sizeof(float) * (size_t)N_NODES * DIM);

    const int TB = 256;
    const int gemm_tiles = (N_NODES / 16) * (WCOLS / 16);

    for (int l = 0; l < NLAYERS; ++l) {
        const float* xsrc = (l == 0) ? x_in
                                     : (out + (size_t)N_NODES * DIM * (size_t)(1 + (l - 1)));
        k_f32_to_f16<<<(N_NODES * DIM + TB - 1) / TB, TB, 0, stream>>>(xsrc, xh, N_NODES * DIM);
        k_pack_weights<<<(WCOLS * DIM + TB - 1) / TB, TB, 0, stream>>>(
            basis + (size_t)l * NBASIS * DIM * DIM, root + (size_t)l * DIM * DIM, wT);
        k_gemm_wmma<<<(gemm_tiles + 7) / 8, TB, 0, stream>>>(xh, wT, x_basis, x_root);
        k_init<<<(N_NODES * DIM + TB - 1) / TB, TB, 0, stream>>>(aggr, nmax, nsum);
        k_edge_logits<<<(N_EDGES + 7) / 8, TB, 0, stream>>>(
            ei, et, att_r + (size_t)l * NREL * NBASIS, att + (size_t)l * 2 * DIM,
            x_basis, x_root, alpha, nmax);
        k_edge_aggr<<<(N_EDGES + 7) / 8, TB, 0, stream>>>(
            ei, et, att_r + (size_t)l * NREL * NBASIS, x_basis, alpha, nmax, nsum, aggr);
        float* state_out = out + (size_t)N_NODES * DIM * (size_t)(1 + l);
        float* final_out = (l == NLAYERS - 1) ? out : nullptr;
        k_node_final<<<(N_NODES + 7) / 8, TB, 0, stream>>>(
            aggr, nsum, x_root, bias + (size_t)l * DIM,
            ln_g + (size_t)l * DIM, ln_b + (size_t)l * DIM, state_out, final_out);
    }
}
